// _ACeDeC_Module_64553358459057
// MI455X (gfx1250) — compile-verified, matrix operand order checked by reading, not run on hardware
//
#include <hip/hip_runtime.h>
#include <math.h>

#define DIM   128
#define K0C   50
#define KOUT  51
#define NPAD  64   // padded output-col space for GEMM2 (51 -> 64)
#define SLABS 4    // 128-row slabs per block (staging amortization)

typedef __attribute__((ext_vector_type(16))) __bf16 v16bf;
typedef __attribute__((ext_vector_type(8)))  __bf16 bf8v;
typedef __attribute__((ext_vector_type(8)))  float  v8f;

// ---------------------------------------------------------------------------
// Prep kernel (1 block, 256 threads): tiny precompute.
//   beta0 = softmax(beta_weights, axis=0)[0]
//   VT[j][d]  = bf16( V[d][j] )                      (B-operand for GEMM1)
//   VWT[k][d] = bf16( sum_d2 V[d][d2] * W[d2][k] )   (B-operand for GEMM2)
//        with W[:,k<50] = beta0*centers0[k], W[:,50] = beta1*centers1[0]
//   cc[k]     = (centers^2) . beta
// ---------------------------------------------------------------------------
__global__ __launch_bounds__(256) void prep_kernel(
    const float* __restrict__ bw, const float* __restrict__ V,
    const float* __restrict__ c0, const float* __restrict__ c1,
    __bf16* __restrict__ VT, __bf16* __restrict__ VWT,
    float* __restrict__ cc, float* __restrict__ beta0g)
{
  __shared__ float b0[DIM], b1[DIM];
  const int t = threadIdx.x;

  if (t < DIM) {
    float e  = __expf(bw[DIM + t] - bw[t]);   // exp(bw1 - bw0)
    float p0 = 1.0f / (1.0f + e);             // softmax over 2 entries
    b0[t] = p0;
    b1[t] = 1.0f - p0;
    beta0g[t] = p0;
  }
  __syncthreads();

  // VT[j*DIM + d] = V[d*DIM + j]
  for (int i = t; i < DIM * DIM; i += 256) {
    int j = i >> 7, d = i & (DIM - 1);
    VT[i] = (__bf16)V[d * DIM + j];
  }

  // cc[k]
  if (t < NPAD) {
    float acc = 0.0f;
    if (t < K0C) {
      for (int d = 0; d < DIM; ++d) { float c = c0[t * DIM + d]; acc += b0[d] * c * c; }
    } else if (t == K0C) {
      for (int d = 0; d < DIM; ++d) { float c = c1[d]; acc += b1[d] * c * c; }
    }
    cc[t] = acc;
  }

  // VWT[k][d] : thread t -> k = t>>2, d in [(t&3)*32, +32)
  {
    const int k = t >> 2;
    const int dbase = (t & 3) * 32;
    for (int d = dbase; d < dbase + 32; ++d) {
      float acc = 0.0f;
      if (k < K0C) {
        for (int d2 = 0; d2 < DIM; ++d2)
          acc += V[d * DIM + d2] * (b0[d2] * c0[k * DIM + d2]);
      } else if (k == K0C) {
        for (int d2 = 0; d2 < DIM; ++d2)
          acc += V[d * DIM + d2] * (b1[d2] * c1[d2]);
      }
      VWT[k * DIM + d] = (__bf16)acc;
    }
  }
}

// ---------------------------------------------------------------------------
// Main kernel: 256 threads = 8 wave32, 16 rows per wave, 128 rows per slab,
// SLABS slabs per block (B-matrices staged into LDS once per block).
// __launch_bounds__(256, 4): cap <=256 VGPRs/wave -> no VGPR-MSB banking,
// occupancy-based latency hiding. __syncthreads() between slabs fences
// cross-slab pipelining (prefetch hints carry next slab's HBM latency).
// GEMM1: z(16x128) @ V(128x128) -> z_rot accumulators -> weighted norms only.
// GEMM2: z(16x128) @ VW(128x64) -> cross; epilogue zz + cc - 2*cross.
// ---------------------------------------------------------------------------
__global__ __launch_bounds__(256, 4) void enrc_kernel(
    const float*  __restrict__ z,
    const __bf16* __restrict__ VTg,
    const __bf16* __restrict__ VWTg,
    const float*  __restrict__ ccg,
    const float*  __restrict__ b0g,
    float*        __restrict__ out,
    int Bn)
{
  __shared__ __align__(16) __bf16 sVT[DIM * DIM];    // 32 KB
  __shared__ __align__(16) __bf16 sVWT[NPAD * DIM];  // 16 KB
  __shared__ float sB0[DIM];
  __shared__ float sCC[NPAD];

  const int t = threadIdx.x;

  // --- cooperative stage of B-matrices into LDS (b128 copies), once/block ---
  {
    const int4* src = (const int4*)VTg;
    int4*       dst = (int4*)sVT;
    #pragma unroll
    for (int i = 0; i < 8; ++i) dst[t + 256 * i] = src[t + 256 * i];   // 2048 int4
    src = (const int4*)VWTg;
    dst = (int4*)sVWT;
    #pragma unroll
    for (int i = 0; i < 4; ++i) dst[t + 256 * i] = src[t + 256 * i];   // 1024 int4
    if (t < DIM)  sB0[t] = b0g[t];
    if (t < NPAD) sCC[t] = ccg[t];
  }
  __syncthreads();

  const int wave = t >> 5;
  const int lane = t & 31;
  const int lr   = lane & 15;   // row within A fragment / col within B,C,D
  const int hh   = lane >> 4;   // half-wave: selects K sub-blocks / row offset
  const int blockRow0 = blockIdx.x * (128 * SLABS);

  // B fragment from LDS (matrix stored transposed: [col][K], contiguous K)
  auto loadB = [&](const __bf16* base, int ntile, int kc) -> v16bf {
    const __bf16* p = base + ((ntile * 16 + lr) * DIM + kc * 32 + hh * 8);
    bf8v lo = *(const bf8v*)p;          // ds_load_b128
    bf8v hi = *(const bf8v*)(p + 16);   // ds_load_b128
    v16bf b;
    #pragma unroll
    for (int i = 0; i < 8; ++i) { b[i] = lo[i]; b[i + 8] = hi[i]; }
    return b;
  };

  #pragma unroll 1
  for (int s = 0; s < SLABS; ++s) {
    const int rowBase = blockRow0 + s * 128 + wave * 16;

    // --- load A fragments: z rows in bf16, ISA 16-bit A 16x32 layout ---
    // lane<16 : row=lr, elems0-7 = K kc*32+0..7,   elems8-15 = K kc*32+16..23
    // lane>=16: row=lr, elems0-7 = K kc*32+8..15,  elems8-15 = K kc*32+24..31
    int arow = rowBase + lr; if (arow >= Bn) arow = Bn - 1;
    const float* zrow = z + (size_t)arow * DIM;

    v16bf A[4];
    #pragma unroll
    for (int kc = 0; kc < 4; ++kc) {
      const int cbase = kc * 32 + hh * 8;
      float4 f0 = *(const float4*)(zrow + cbase);
      float4 f1 = *(const float4*)(zrow + cbase + 4);
      float4 f2 = *(const float4*)(zrow + cbase + 16);
      float4 f3 = *(const float4*)(zrow + cbase + 20);
      v16bf a;
      a[0]  = (__bf16)f0.x; a[1]  = (__bf16)f0.y; a[2]  = (__bf16)f0.z; a[3]  = (__bf16)f0.w;
      a[4]  = (__bf16)f1.x; a[5]  = (__bf16)f1.y; a[6]  = (__bf16)f1.z; a[7]  = (__bf16)f1.w;
      a[8]  = (__bf16)f2.x; a[9]  = (__bf16)f2.y; a[10] = (__bf16)f2.z; a[11] = (__bf16)f2.w;
      a[12] = (__bf16)f3.x; a[13] = (__bf16)f3.y; a[14] = (__bf16)f3.z; a[15] = (__bf16)f3.w;
      A[kc] = a;
    }

    // prefetch next slab's z row for this lane (cache-side, no registers held)
    if (s + 1 < SLABS) {
      int nrow = arow + 128; if (nrow >= Bn) nrow = Bn - 1;
      __builtin_prefetch(z + (size_t)nrow * DIM, 0, 3);
      __builtin_prefetch(z + (size_t)nrow * DIM + 64, 0, 3);
    }

    // --- GEMM1: z @ V, reduce to weighted norms only ---
    float zz0p[8], sp[8];
    #pragma unroll
    for (int v = 0; v < 8; ++v) { zz0p[v] = 0.0f; sp[v] = 0.0f; }

    for (int nt = 0; nt < 8; ++nt) {
      v8f acc = {};
      #pragma unroll
      for (int kc = 0; kc < 4; ++kc) {
        v16bf b = loadB(sVT, nt, kc);
        acc = __builtin_amdgcn_wmma_f32_16x16x32_bf16(
                  false, A[kc], false, b, (short)0, acc, false, false);
      }
      const float w = sB0[nt * 16 + lr];  // beta0[col] for all 8 elems of lane
      #pragma unroll
      for (int v = 0; v < 8; ++v) {
        float x = acc[v];
        sp[v]   += x * x;
        zz0p[v] += w * x * x;
      }
    }

    // reduce over the 16 lanes holding the same rows (lane bits 0..3)
    #pragma unroll
    for (int v = 0; v < 8; ++v) {
      #pragma unroll
      for (int m = 1; m < 16; m <<= 1) {
        zz0p[v] += __shfl_xor(zz0p[v], m, 32);
        sp[v]   += __shfl_xor(sp[v],   m, 32);
      }
    }
    // now lane's zz0p[v]/sp[v] hold the norm of row (v + hh*8) -- exactly the
    // row its C/D accumulator element v corresponds to.

    // --- GEMM2: z @ (V @ W) = cross, fused epilogue ---
    for (int nt = 0; nt < 4; ++nt) {
      v8f acc = {};
      #pragma unroll
      for (int kc = 0; kc < 4; ++kc) {
        v16bf b = loadB(sVWT, nt, kc);
        acc = __builtin_amdgcn_wmma_f32_16x16x32_bf16(
                  false, A[kc], false, b, (short)0, acc, false, false);
      }
      const int col = nt * 16 + lr;
      if (col < KOUT) {
        const float ccv = sCC[col];
        #pragma unroll
        for (int v = 0; v < 8; ++v) {
          const int r = rowBase + v + hh * 8;
          if (r < Bn) {
            float zz = (col < K0C) ? zz0p[v] : (sp[v] - zz0p[v]);
            out[(size_t)r * KOUT + col] = zz + ccv - 2.0f * acc[v];
          }
        }
      }
    }

    // fence: keep per-slab register lifetimes separate (uniform loop -> safe)
    __syncthreads();
  }
}

// ---------------------------------------------------------------------------
extern "C" void kernel_launch(void* const* d_in, const int* in_sizes, int n_in,
                              void* d_out, int out_size, void* d_ws, size_t ws_size,
                              hipStream_t stream)
{
  const float* z  = (const float*)d_in[0];   // [B,128]
  const float* bw = (const float*)d_in[1];   // [2,128]
  const float* V  = (const float*)d_in[2];   // [128,128]
  const float* c0 = (const float*)d_in[3];   // [50,128]
  const float* c1 = (const float*)d_in[4];   // [1,128]
  float* out = (float*)d_out;                // [B,51]

  char* ws = (char*)d_ws;
  __bf16* VT  = (__bf16*)(ws);               // 32768 B
  __bf16* VWT = (__bf16*)(ws + 32768);       // 16384 B
  float*  cc  = (float*)(ws + 49152);        //   256 B
  float*  b0  = (float*)(ws + 49408);        //   512 B

  const int Bn = in_sizes[0] / DIM;          // 262144

  prep_kernel<<<1, 256, 0, stream>>>(bw, V, c0, c1, VT, VWT, cc, b0);

  const int rowsPerBlock = 128 * SLABS;
  const int blocks = (Bn + rowsPerBlock - 1) / rowsPerBlock;   // 512
  enrc_kernel<<<blocks, 256, 0, stream>>>(z, VT, VWT, cc, b0, out, Bn);
}
